// Model_NoLocalSocial_LSTM_F_3796751089883
// MI455X (gfx1250) — matrix operand order repaired and checked by reading, not compiled, not run on hardware
//
#include <hip/hip_runtime.h>
#include <hip/hip_bf16.h>
#include <cstddef>
#include <cstdint>

typedef _Float16 h16;
typedef __attribute__((ext_vector_type(16))) _Float16 v16h;
typedef __attribute__((ext_vector_type(8)))  _Float16 v8h;
typedef __attribute__((ext_vector_type(8)))  float    v8f;

#define SOC_H 13
#define ATT_N 5
#define ATT_D 32
#define HID 64
#define DEC_H 128
#define PRED_T 25
#define BATCH 4096
#define SEQ 32
#define ENC_M (BATCH*5)        // 20480
#define NPOS 33                // conv output positions 11*3
#define CONV_K 384             // fused K dim: 6 taps * 64

// ---------------------------------------------------------------- helpers ----
__device__ __forceinline__ int lane_id() { return threadIdx.x & 31; }

__device__ __forceinline__ float fast_tanh(float x) {
#if __has_builtin(__builtin_amdgcn_tanhf)
  return __builtin_amdgcn_tanhf(x);   // v_tanh_f32 (gfx1250 TRANS op)
#else
  return tanhf(x);
#endif
}
__device__ __forceinline__ float sigm(float x) { return 0.5f + 0.5f * fast_tanh(0.5f * x); }
__device__ __forceinline__ float lrelu(float x) { return x > 0.f ? x : 0.1f * x; }

// A fragment (16x32 f16) from row-major tile base (rows m..m+15), ld in halves.
// Lanes 0-15: row=l, K = {k0+0..7, k0+16..23}; lanes 16-31: row=l-16, K={k0+8..15, k0+24..31}
__device__ __forceinline__ v16h load_a_frag(const h16* src, int ld, int k0) {
  int l = lane_id();
  const h16* p0 = src + (size_t)(l & 15) * ld + k0 + ((l < 16) ? 0 : 8);
  v8h a0 = *(const v8h*)p0;         // 16B vector load
  v8h a1 = *(const v8h*)(p0 + 16);  // 16B vector load
  v16h a;
#pragma unroll
  for (int j = 0; j < 8; ++j) { a[j] = a0[j]; a[8 + j] = a1[j]; }
  return a;
}

// B fragment from pre-packed fragment-order buffer:
// packed[((kt*NT + nt)*32 + lane)*16 + j]  -> one contiguous 32B load per lane
__device__ __forceinline__ v16h load_b_packed(const h16* packed, int NT, int kt, int nt) {
  const h16* p = packed + (((size_t)(kt * NT + nt) * 32 + lane_id()) << 4);
  return *(const v16h*)p;
}

__device__ __forceinline__ v8f wmma16(v16h a, v16h b, v8f c) {
  return __builtin_amdgcn_wmma_f32_16x16x32_f16(false, a, false, b, (short)0, c, false, false);
}

// ---------------------------------------------------- weight prep kernels ----
// Pack f32 [N][K] weight (out-features major, as in the model) into f16
// WMMA-B fragment order. Element t encodes (frag, lane, j) directly.
__global__ void k_pack_b(const float* __restrict__ src, h16* __restrict__ dst,
                         int N, int K) {
  int t = blockIdx.x * blockDim.x + threadIdx.x;
  if (t >= N * K) return;
  int j    = t & 15;
  int lane = (t >> 4) & 31;
  int frag = t >> 9;
  int NT = N >> 4;
  int nt = frag % NT;
  int kt = frag / NT;
  int kk = ((lane < 16) ? 0 : 8) + (j & 7) + ((j >> 3) << 4);
  int n = nt * 16 + (lane & 15);
  int k = kt * 32 + kk;
  dst[t] = (h16)src[(size_t)n * K + k];
}

// Fused conv weights, written directly in packed fragment order (NT=10):
// FK(k=tap*64+d, n) = sum_c Kw[n][c][kh][kw] * Wk[c][d]
__global__ void k_fuse_convw(const float* __restrict__ Kw, const float* __restrict__ Wk,
                             const float* __restrict__ Vw, const float* __restrict__ Wv,
                             h16* __restrict__ FK, h16* __restrict__ FV) {
  int t = blockIdx.x * blockDim.x + threadIdx.x;
  const int per = 6 * 64 * 160;
  if (t >= 2 * per) return;
  int which = t >= per;
  int u = t % per;
  int tap = u / (64 * 160);
  int rem = u % (64 * 160);
  int d = rem / 160, n = rem % 160;
  int kh = tap >> 1, kw = tap & 1;
  const float* Cw = which ? Vw : Kw;
  const float* Lw = which ? Wv : Wk;
  float s = 0.f;
  for (int c = 0; c < 160; ++c)
    s += Cw[(((size_t)n * 160 + c) * 3 + kh) * 2 + kw] * Lw[(size_t)c * 64 + d];
  int k = tap * 64 + d;
  int kt = k >> 5, kk = k & 31;
  int lane = (n & 15) + (((kk >> 3) & 1) ? 16 : 0);
  int j = (kk & 7) + ((kk >> 4) << 3);
  int nt = n >> 4;
  size_t idx = (((size_t)(kt * 10 + nt) * 32 + lane) << 4) + j;
  (which ? FV : FK)[idx] = (h16)s;
}

// bK[ow][n] = Kb[n] + sum_{kh, kw in-range(ow)} sum_c Kw[n][c][kh][kw]*bk[c]
__global__ void k_conv_bias(const float* __restrict__ Kw, const float* __restrict__ Kb,
                            const float* __restrict__ bk,
                            const float* __restrict__ Vw, const float* __restrict__ Vb,
                            const float* __restrict__ bv,
                            float* __restrict__ bKo, float* __restrict__ bVo) {
  int t = blockIdx.x * blockDim.x + threadIdx.x;
  if (t >= 2 * 3 * 160) return;
  int which = t >= 3 * 160;
  int u = t % (3 * 160);
  int ow = u / 160, n = u % 160;
  const float* Cw = which ? Vw : Kw;
  const float* cb = which ? Vb : Kb;
  const float* lb = which ? bv : bk;
  float s = cb[n];
  for (int kh = 0; kh < 3; ++kh)
    for (int kw = 0; kw < 2; ++kw) {
      int iw = ow + kw - 1;
      if (iw >= 0 && iw <= 1)
        for (int c = 0; c < 160; ++c)
          s += Cw[(((size_t)n * 160 + c) * 3 + kh) * 2 + kw] * lb[c];
    }
  (which ? bVo : bKo)[(size_t)ow * 160 + n] = s;
}

// ---------------------------------------------------------- encoder LSTM ----
// 1280 blocks x 128 threads (4 waves). Wave w owns gate w's 64 columns.
__global__ __launch_bounds__(128) void k_enc_lstm(
    const float* __restrict__ hist_tags, const float* __restrict__ hist_nghs,
    const float* __restrict__ W_emb, const float* __restrict__ b_emb,
    const h16* __restrict__ WihP, const h16* __restrict__ WhhP,
    const float* __restrict__ enc_b, float* __restrict__ enc_out /*[M][64]*/) {
  __shared__ __align__(16) h16  xA[16][32];
  __shared__ __align__(16) h16  hA[16][64];
  __shared__ float gates[16][256];
  __shared__ float cS[16][64];
  __shared__ float hF[16][64];

  const int m0 = blockIdx.x * 16;
  const int tid = threadIdx.x;
  const int wave = tid >> 5;
  const int l = tid & 31;

  for (int e = tid; e < 16 * 64; e += 128) {
    int m = e >> 6, n = e & 63;
    hA[m][n] = (h16)0.f; cS[m][n] = 0.f;
  }

  // resident B fragments (packed, NT=16)
  v16h Bx[4];
  v16h Bh[4][2];
#pragma unroll
  for (int nt = 0; nt < 4; ++nt) {
    int nti = wave * 4 + nt;
    Bx[nt]    = load_b_packed(WihP, 16, 0, nti);
    Bh[nt][0] = load_b_packed(WhhP, 16, 0, nti);
    Bh[nt][1] = load_b_packed(WhhP, 16, 1, nti);
  }
  __syncthreads();

  for (int s = 0; s < SEQ; ++s) {
    // fused embedding: x = lrelu(in @ W_emb.T + b_emb) for the 16x32 tile
    for (int e = tid; e < 512; e += 128) {
      int m = e >> 5, k = e & 31;
      int row = m0 + m;
      int b = row / 5, a = row % 5;
      const float* in = (a == 0)
          ? hist_tags + ((size_t)b * SEQ + s) * 6
          : hist_nghs + (((size_t)b * 4 + (a - 1)) * SEQ + s) * 6;
      float v = b_emb[k];
#pragma unroll
      for (int j = 0; j < 6; ++j) v += in[j] * W_emb[k * 6 + j];
      xA[m][k] = (h16)lrelu(v);
    }
    __syncthreads();

#pragma unroll
    for (int nt = 0; nt < 4; ++nt) {
      int n0 = wave * 64 + nt * 16;
      float bias = enc_b[n0 + (l & 15)];
      v8f acc;
#pragma unroll
      for (int r = 0; r < 8; ++r) acc[r] = bias;
      v16h a0 = load_a_frag(&xA[0][0], 32, 0);
      acc = wmma16(a0, Bx[nt], acc);
      v16h a1 = load_a_frag(&hA[0][0], 64, 0);
      acc = wmma16(a1, Bh[nt][0], acc);
      v16h a2 = load_a_frag(&hA[0][0], 64, 32);
      acc = wmma16(a2, Bh[nt][1], acc);
      int col = n0 + (l & 15);
      int rb = (l >> 4) * 8;
#pragma unroll
      for (int r = 0; r < 8; ++r) gates[rb + r][col] = acc[r];
    }
    __syncthreads();

    for (int e = tid; e < 1024; e += 128) {
      int m = e >> 6, n = e & 63;
      float gi = gates[m][n],       gf = gates[m][64 + n];
      float gg = gates[m][128 + n], go = gates[m][192 + n];
      float c = sigm(gf) * cS[m][n] + sigm(gi) * fast_tanh(gg);
      float h = sigm(go) * fast_tanh(c);
      cS[m][n] = c; hA[m][n] = (h16)h; hF[m][n] = h;
    }
    __syncthreads();
  }

  for (int e = tid; e < 1024; e += 128) {
    int m = e >> 6, n = e & 63;
    enc_out[(size_t)(m0 + m) * 64 + n] = hF[m][n];
  }
}

// ---------------------------------------------------------- social scatter ----
__global__ void k_social_grid(const float* __restrict__ pose_tags,
                              const float* __restrict__ pose_nghs,
                              const float* __restrict__ enc_out,
                              h16* __restrict__ sm /*[B][13][2][64]*/) {
  int b = blockIdx.x;
  int t = threadIdx.x;  // 64
  h16* smb = sm + (size_t)b * 26 * 64;
  for (int c = 0; c < 26; ++c) smb[c * 64 + t] = (h16)0.f;
  float px = pose_tags[((size_t)b * SEQ + 31) * 2 + 0];
  for (int j = 0; j < 4; ++j) {
    const float* pn = pose_nghs + (((size_t)b * 4 + j) * SEQ + 31) * 2;
    float x = pn[0], y = pn[1];
    bool skip = fabsf(x + y) < 1e-4f;
    int ix = (SOC_H - 1) / 2 + (int)rintf((x - px) * 0.25f);
    int iy = (y >= 0.f) ? 1 : 0;
    if (!skip && ix >= 0 && ix < SOC_H)
      smb[(ix * 2 + iy) * 64 + t] = (h16)enc_out[((size_t)b * 5 + 1 + j) * 64 + t];
  }
}

// ----------------------------------------------- fused linear+conv -> K,V ----
// Rows r = b*33 + (oh*3+ow). A 16-row tile touches at most 2 consecutive b's;
// stage both social grids into LDS via CDNA5 async global->LDS, then feed
// A-fragments from LDS. 4 waves x 5 N-tiles cover combined 320 cols [K|V].
__device__ __forceinline__ v16h load_a_conv(const h16* __restrict__ smT,
                                            int b_lo, int m0, int kt) {
  int l = lane_id();
  int row = m0 + (l & 15);
  int b = row / NPOS, p = row % NPOS;
  int oh = p / 3, ow = p % 3;
  int tap = kt >> 1;
  int kh = tap >> 1, kw = tap & 1;
  int ih = oh + kh, iw = ow + kw - 1;
  int kb = (l < 16) ? 0 : 8;
  int d0 = (kt & 1) * 32 + kb;
  v16h a;
  if (iw < 0 || iw > 1) {
#pragma unroll
    for (int j = 0; j < 16; ++j) a[j] = (h16)0.f;
  } else {
    const h16* p0 = smT + (((b - b_lo) * 26 + ih * 2 + iw) * 64 + d0);
    v8h a0 = *(const v8h*)p0;
    v8h a1 = *(const v8h*)(p0 + 16);
#pragma unroll
    for (int j = 0; j < 8; ++j) { a[j] = a0[j]; a[8 + j] = a1[j]; }
  }
  return a;
}

__global__ __launch_bounds__(128) void k_conv_kv(
    const h16* __restrict__ sm, const h16* __restrict__ FK, const h16* __restrict__ FV,
    const float* __restrict__ bK, const float* __restrict__ bV,
    h16* __restrict__ karr /*[B][33][160]*/, h16* __restrict__ varr) {
  __shared__ __align__(16) h16 smT[2 * 26 * 64];

  const int m0 = blockIdx.x * 16;
  const int wave = threadIdx.x >> 5;
  const int l = threadIdx.x & 31;
  const int b_lo = m0 / NPOS;

  // async-stage the (up to) two social grids this tile touches: 2 x 3328 B
  uint32_t lds_base = (uint32_t)(uintptr_t)(&smT[0]);
#pragma unroll
  for (int half = 0; half < 2; ++half) {
    int bsrc = b_lo + half; if (bsrc > BATCH - 1) bsrc = BATCH - 1;
    const h16* src = sm + (size_t)bsrc * (26 * 64);
    for (int idx = threadIdx.x; idx < 208; idx += 128) {     // 208 x 16B = 3328B
      uint64_t ga = (uint64_t)(uintptr_t)(src + idx * 8);
      uint32_t la = lds_base + (uint32_t)(half * 3328 + idx * 16);
      asm volatile("global_load_async_to_lds_b128 %0, %1, off"
                   :: "v"(la), "v"(ga) : "memory");
    }
  }
  asm volatile("s_wait_asynccnt 0x0" ::: "memory");
  __syncthreads();

  v8f acc[5];
#pragma unroll
  for (int i = 0; i < 5; ++i)
#pragma unroll
    for (int r = 0; r < 8; ++r) acc[i][r] = 0.f;

  for (int kt = 0; kt < CONV_K / 32; ++kt) {
    v16h a = load_a_conv(smT, b_lo, m0, kt);
#pragma unroll
    for (int i = 0; i < 5; ++i) {
      int n0c = wave * 80 + i * 16;
      const h16* W = (n0c < 160) ? FK : FV;
      int nt = ((n0c < 160) ? n0c : n0c - 160) >> 4;
      v16h bf = load_b_packed(W, 10, kt, nt);
      acc[i] = wmma16(a, bf, acc[i]);
    }
  }
#pragma unroll
  for (int i = 0; i < 5; ++i) {
    int n0c = wave * 80 + i * 16;
    bool isV = n0c >= 160;
    int n = (isV ? n0c - 160 : n0c) + (l & 15);
    const float* bias = isV ? bV : bK;
    h16* dst = isV ? varr : karr;
    int rb = (l >> 4) * 8;
#pragma unroll
    for (int r = 0; r < 8; ++r) {
      int row = m0 + rb + r;
      int b = row / NPOS, p = row % NPOS, ow = p % 3;
      dst[((size_t)b * NPOS + p) * 160 + n] = (h16)(acc[i][r] + bias[ow * 160 + n]);
    }
  }
}

// --------------------------------------------------------------- attention ----
__global__ void k_attention(const float* __restrict__ enc_out,
                            const float* __restrict__ Wq, const float* __restrict__ bq,
                            const h16* __restrict__ karr, const h16* __restrict__ varr,
                            float* __restrict__ nls /*[B][160]*/) {
  int t = blockIdx.x * blockDim.x + threadIdx.x;
  if (t >= BATCH * ATT_N) return;
  int b = t / ATT_N, h = t % ATT_N;
  const float* e0 = enc_out + (size_t)b * 5 * 64;
  float q[ATT_D];
  for (int d = 0; d < ATT_D; ++d) {
    float v = bq[h * ATT_D + d];
    const float* w = Wq + (size_t)(h * ATT_D + d) * 64;
    for (int k = 0; k < 64; ++k) v += e0[k] * w[k];
    q[d] = v;
  }
  float logit[NPOS];
  float mx = -1e30f;
  const h16* kb_ = karr + (size_t)b * NPOS * 160 + h * ATT_D;
  for (int p = 0; p < NPOS; ++p) {
    const h16* kp = kb_ + (size_t)p * 160;
    float s = 0.f, ab = 0.f;
    for (int d = 0; d < ATT_D; ++d) { float kv = (float)kp[d]; s += q[d] * kv; ab += fabsf(kv); }
    float lg = (ab < 1e-10f) ? -1e10f : s * 0.17677669529663687f; // 1/sqrt(32)
    logit[p] = lg; mx = fmaxf(mx, lg);
  }
  float den = 0.f;
  for (int p = 0; p < NPOS; ++p) { logit[p] = __expf(logit[p] - mx); den += logit[p]; }
  float inv = __builtin_amdgcn_rcpf(den);
  const h16* vb_ = varr + (size_t)b * NPOS * 160 + h * ATT_D;
  for (int d = 0; d < ATT_D; ++d) {
    float s = 0.f;
    for (int p = 0; p < NPOS; ++p) s += logit[p] * (float)vb_[(size_t)p * 160 + d];
    nls[(size_t)b * 160 + h * ATT_D + d] = s * inv;
  }
}

// -------------------------------------------------- decoder input build ----
__global__ void k_dec_input(const float* __restrict__ enc_out, const float* __restrict__ nls,
                            const float* __restrict__ W_fcn, const float* __restrict__ b_fcn,
                            h16* __restrict__ dec_inp /*[B][128]*/) {
  int t = blockIdx.x * blockDim.x + threadIdx.x;
  if (t >= BATCH * 128) return;
  int b = t >> 7, j = t & 127;
  float v;
  if (j < 64) {
    v = enc_out[(size_t)b * 5 * 64 + j];
  } else {
    int o = j - 64;
    v = b_fcn[o];
    const float* w = W_fcn + (size_t)o * 160;
    const float* x = nls + (size_t)b * 160;
    for (int k = 0; k < 160; ++k) v += x[k] * w[k];
  }
  dec_inp[(size_t)b * 128 + j] = (h16)v;
}

// ------------------------------------ generic WMMA GEMM with bias (f32 C) ----
__global__ __launch_bounds__(32) void k_gemm_bias(
    const h16* __restrict__ A, const h16* __restrict__ B /*packed*/,
    const float* __restrict__ bias, float* __restrict__ C, int N, int K) {
  int mt = blockIdx.x, nt = blockIdx.y;
  int l = threadIdx.x & 31;
  int n0 = nt * 16;
  float bv = bias[n0 + (l & 15)];
  v8f acc;
#pragma unroll
  for (int r = 0; r < 8; ++r) acc[r] = bv;
  const h16* Ab = A + (size_t)mt * 16 * K;
  for (int k0 = 0; k0 < K; k0 += 32) {
    v16h a = load_a_frag(Ab, K, k0);
    v16h bf = load_b_packed(B, N >> 4, k0 >> 5, nt);
    acc = wmma16(a, bf, acc);
  }
  int col = n0 + (l & 15);
  int rb = (l >> 4) * 8;
#pragma unroll
  for (int r = 0; r < 8; ++r) C[(size_t)(mt * 16 + rb + r) * N + col] = acc[r];
}

// ---------------------------------------------------------- decoder LSTM ----
// 256 blocks x 256 threads (8 waves). Wave w: gate columns [64w, 64w+64).
// xg = dec_inp @ dec_Wih.T + dec_b (precomputed). Fused 128->2 output head.
__global__ __launch_bounds__(256) void k_dec_lstm(
    const h16* __restrict__ WhhP /*packed NT=32*/, const float* __restrict__ xg /*[B][512]*/,
    const float* __restrict__ pose_tags, const float* __restrict__ W_out,
    const float* __restrict__ b_out, float* __restrict__ out /*[B][25][2]*/) {
  __shared__ __align__(16) h16  hA[16][128];
  __shared__ float cS[16][128];
  __shared__ float hF[16][128];
  __shared__ float gates[16][512];

  const int m0 = blockIdx.x * 16;
  const int tid = threadIdx.x;
  const int wave = tid >> 5;
  const int l = tid & 31;

  for (int e = tid; e < 16 * 128; e += 256) {
    int m = e >> 7, n = e & 127;
    hA[m][n] = (h16)0.f; cS[m][n] = 0.f;
  }

  v16h Bw[4][4];
#pragma unroll
  for (int nt = 0; nt < 4; ++nt) {
#pragma unroll
    for (int ks = 0; ks < 4; ++ks)
      Bw[nt][ks] = load_b_packed(WhhP, 32, ks, wave * 4 + nt);
  }
  __syncthreads();

  for (int t = 0; t < PRED_T; ++t) {
#pragma unroll
    for (int nt = 0; nt < 4; ++nt) {
      int n0 = wave * 64 + nt * 16;
      int col = n0 + (l & 15);
      int rb = (l >> 4) * 8;
      v8f acc;
#pragma unroll
      for (int r = 0; r < 8; ++r) acc[r] = xg[(size_t)(m0 + rb + r) * 512 + col];
#pragma unroll
      for (int ks = 0; ks < 4; ++ks) {
        v16h a = load_a_frag(&hA[0][0], 128, ks * 32);
        acc = wmma16(a, Bw[nt][ks], acc);
      }
#pragma unroll
      for (int r = 0; r < 8; ++r) gates[rb + r][col] = acc[r];
    }
    __syncthreads();

    for (int e = tid; e < 2048; e += 256) {
      int m = e >> 7, n = e & 127;
      float gi = gates[m][n],        gf = gates[m][128 + n];
      float gg = gates[m][256 + n],  go = gates[m][384 + n];
      float c = sigm(gf) * cS[m][n] + sigm(gi) * fast_tanh(gg);
      float h = sigm(go) * fast_tanh(c);
      cS[m][n] = c; hA[m][n] = (h16)h; hF[m][n] = h;
    }
    __syncthreads();

    if (tid < 32) {
      int m = tid >> 1, ch = tid & 1;
      float s = b_out[ch];
      for (int n = 0; n < 128; ++n) s += lrelu(hF[m][n]) * W_out[ch * 128 + n];
      int b = m0 + m;
      out[((size_t)b * PRED_T + t) * 2 + ch] =
          pose_tags[((size_t)b * SEQ + 31) * 2 + ch] + s;
    }
    __syncthreads();
  }
}

// ------------------------------------------------------------------ launch ----
extern "C" void kernel_launch(void* const* d_in, const int* in_sizes, int n_in,
                              void* d_out, int out_size, void* d_ws, size_t ws_size,
                              hipStream_t stream) {
  const float* pose_tags = (const float*)d_in[0];
  const float* hist_tags = (const float*)d_in[1];
  const float* pose_nghs = (const float*)d_in[2];
  const float* hist_nghs = (const float*)d_in[3];
  const float* W_emb   = (const float*)d_in[4];
  const float* b_emb   = (const float*)d_in[5];
  const float* enc_Wih = (const float*)d_in[6];
  const float* enc_Whh = (const float*)d_in[7];
  const float* enc_b   = (const float*)d_in[8];
  const float* Wq  = (const float*)d_in[9];
  const float* bq  = (const float*)d_in[10];
  const float* Wk  = (const float*)d_in[11];
  const float* bk  = (const float*)d_in[12];
  const float* Wv  = (const float*)d_in[13];
  const float* bv  = (const float*)d_in[14];
  const float* Kw  = (const float*)d_in[15];
  const float* Kb  = (const float*)d_in[16];
  const float* Vw  = (const float*)d_in[17];
  const float* Vb  = (const float*)d_in[18];
  const float* W_fcn = (const float*)d_in[19];
  const float* b_fcn = (const float*)d_in[20];
  const float* dec_Wih = (const float*)d_in[21];
  const float* dec_Whh = (const float*)d_in[22];
  const float* dec_b   = (const float*)d_in[23];
  const float* W_out = (const float*)d_in[24];
  const float* b_out = (const float*)d_in[25];
  float* out = (float*)d_out;

  char* ws = (char*)d_ws;
  size_t off = 0;
  auto take = [&](size_t bytes) -> char* {
    char* p = ws + off;
    off = (off + bytes + 255) & ~(size_t)255;
    return p;
  };
  float* enc_out = (float*)take((size_t)ENC_M * 64 * 4);
  h16*   sm      = (h16*)  take((size_t)BATCH * 26 * 64 * 2);
  h16*   karr    = (h16*)  take((size_t)BATCH * NPOS * 160 * 2);
  h16*   varr    = (h16*)  take((size_t)BATCH * NPOS * 160 * 2);
  float* nls     = (float*)take((size_t)BATCH * 160 * 4);
  h16*   dec_inp = (h16*)  take((size_t)BATCH * 128 * 2);
  float* xg      = (float*)take((size_t)BATCH * 512 * 4);
  h16* encWihP = (h16*)take((size_t)32 * 256 * 2);
  h16* encWhhP = (h16*)take((size_t)64 * 256 * 2);
  h16* decWihP = (h16*)take((size_t)128 * 512 * 2);
  h16* decWhhP = (h16*)take((size_t)128 * 512 * 2);
  h16* FK = (h16*)take((size_t)CONV_K * 160 * 2);
  h16* FV = (h16*)take((size_t)CONV_K * 160 * 2);
  float* bKo = (float*)take((size_t)3 * 160 * 4);
  float* bVo = (float*)take((size_t)3 * 160 * 4);
  (void)ws_size; (void)n_in; (void)in_sizes; (void)out_size;

  // weight prep (pack into WMMA-B fragment order)
  k_pack_b<<<(256 * 32 + 255) / 256, 256, 0, stream>>>(enc_Wih, encWihP, 256, 32);
  k_pack_b<<<(256 * 64 + 255) / 256, 256, 0, stream>>>(enc_Whh, encWhhP, 256, 64);
  k_pack_b<<<(512 * 128 + 255) / 256, 256, 0, stream>>>(dec_Wih, decWihP, 512, 128);
  k_pack_b<<<(512 * 128 + 255) / 256, 256, 0, stream>>>(dec_Whh, decWhhP, 512, 128);
  k_fuse_convw<<<(2 * 6 * 64 * 160 + 255) / 256, 256, 0, stream>>>(Kw, Wk, Vw, Wv, FK, FV);
  k_conv_bias<<<(2 * 3 * 160 + 255) / 256, 256, 0, stream>>>(Kw, Kb, bk, Vw, Vb, bv, bKo, bVo);

  // encoder
  k_enc_lstm<<<ENC_M / 16, 128, 0, stream>>>(hist_tags, hist_nghs, W_emb, b_emb,
                                             encWihP, encWhhP, enc_b, enc_out);
  // social grid + K/V + attention
  k_social_grid<<<BATCH, 64, 0, stream>>>(pose_tags, pose_nghs, enc_out, sm);
  k_conv_kv<<<(BATCH * NPOS) / 16, 128, 0, stream>>>(sm, FK, FV, bKo, bVo, karr, varr);
  k_attention<<<(BATCH * ATT_N + 255) / 256, 256, 0, stream>>>(enc_out, Wq, bq, karr, varr, nls);

  // decoder
  k_dec_input<<<(BATCH * 128 + 255) / 256, 256, 0, stream>>>(enc_out, nls, W_fcn, b_fcn, dec_inp);
  {
    dim3 g(BATCH / 16, 512 / 16);
    k_gemm_bias<<<g, 32, 0, stream>>>(dec_inp, decWihP, dec_b, xg, 512, 128);
  }
  k_dec_lstm<<<BATCH / 16, 256, 0, stream>>>(decWhhP, xg, pose_tags, W_out, b_out, out);
}
